// Encoder_20126216749963
// MI455X (gfx1250) — matrix operand order repaired and checked by reading, not compiled, not run on hardware
//
#include <hip/hip_runtime.h>
#include <hip/hip_bf16.h>
#include <stdint.h>

// Problem constants
#define HSZ 1200           // hidden size
#define HP  1216           // hidden padded to multiple of 32 (WMMA K granularity)
#define BSZ 64             // batch
#define TSZ 512            // sequence length
#define ISZ 30             // input features (layer 0)
#define IP  32             // padded input features
#define G3  3600           // 3*H
#define LP  1224           // LDS row pitch (HP + 8 bf16 = +16B) -> conflict-free ds_load_b128
#define NL0 75             // layer-0 blocks (75 x 16 = 1200 columns)
#define NBLK 150           // total persistent blocks (75 per layer)

typedef __attribute__((ext_vector_type(16))) __bf16 v16bf;
typedef __attribute__((ext_vector_type(8)))  float  v8f;

__device__ __forceinline__ float fast_sigmoid(float x) {
    return 1.0f / (1.0f + __expf(-x));
}
__device__ __forceinline__ float fast_tanh(float x) {
    return 2.0f / (1.0f + __expf(-2.0f * x)) - 1.0f;
}

// A-operand tile: 16 contiguous K elems, p = row_base + k0 + half*16
__device__ __forceinline__ v16bf load_a_tile(const __bf16* __restrict__ p) {
    v16bf r;
#pragma unroll
    for (int i = 0; i < 8; ++i) r[i] = p[i];
#pragma unroll
    for (int i = 0; i < 8; ++i) r[8 + i] = p[8 + i];
    return r;
}
// B-operand tile (same contiguous k order as A): 16 contiguous K elems, p = row_base + k0 + half*16
__device__ __forceinline__ v16bf load_b_tile(const __bf16* p) {
    v16bf r;
#pragma unroll
    for (int i = 0; i < 16; ++i) r[i] = p[i];
    return r;
}

__device__ __forceinline__ v8f wmma_bf16(v16bf a, v16bf b, v8f c) {
    return __builtin_amdgcn_wmma_f32_16x16x32_bf16(
        false, a, false, b, (short)0, c, false, false);
}

// One fused [16 rows] x [3 gate column tiles] GEMM slice:
// A from global (row pitch = Kp implicit in aRow addressing), B tiles from LDS.
__device__ __forceinline__ void gemm3(const __bf16* __restrict__ aRow,
                                      const __bf16* bR, const __bf16* bZ, const __bf16* bN,
                                      int K, v8f& accR, v8f& accZ, v8f& accN)
{
#pragma unroll 2
    for (int k0 = 0; k0 < K; k0 += 32) {
        v16bf a = load_a_tile(aRow + k0);
        accR = wmma_bf16(a, load_b_tile(bR + k0), accR);
        accZ = wmma_bf16(a, load_b_tile(bZ + k0), accZ);
        accN = wmma_bf16(a, load_b_tile(bN + k0), accN);
    }
}

struct GruParams {
    const __bf16* xb;                         // [T][64][IP]
    const __bf16* Wih0;                       // [3600][IP]
    const __bf16* Whh0;                       // [3600][HP]
    const __bf16* Wih1;                       // [3600][HP]
    const __bf16* Whh1;                       // [3600][HP]
    const float *b_ih0, *b_hh0, *b_ih1, *b_hh1;
    __bf16* h0b[2]; float* h0f[2];
    __bf16* h1b[2]; float* h1f[2];
    unsigned* bar;                            // zero-initialized epoch counter
};

// Persistent two-layer GRU. Blocks 0..74: layer 0 (step k); blocks 75..149: layer 1 (step k-1).
// One device-wide barrier per iteration; weights LDS-resident for all 512 steps.
__global__ __launch_bounds__(128)
void gru_persistent(GruParams P)
{
    extern __shared__ __align__(16) char smem_raw[];
    __bf16* sW = (__bf16*)smem_raw;

    const bool isL0 = blockIdx.x < NL0;
    const int  colb = isL0 ? blockIdx.x : (blockIdx.x - NL0);
    const int  col  = colb * 16;
    const int  tid  = threadIdx.x;
    const int  lane = tid & 31;
    const int  wave = tid >> 5;
    const int  half = lane >> 4;
    const int  lr   = lane & 15;
    const int  m0   = wave * 16;
    const int  cn   = col + lr;

    // LDS layout: [0 .. 48*LP)     = Whh tiles (3 gates x 16 rows, pitch LP)
    //             [48*LP .. )      = Wih tiles (pitch LP for layer1, pitch IP for layer0)
    __bf16* sWhh = sW;
    __bf16* sWih = sW + 48 * LP;
    const int ihPitch = isL0 ? IP : LP;

    // ---- one-time weight staging into LDS (reused 512x) ----
    {
        const __bf16* gWhh = isL0 ? P.Whh0 : P.Whh1;
        const int cpr = HP / 8;                       // 16B chunks per row
        for (int idx = tid; idx < 48 * cpr; idx += blockDim.x) {
            int row = idx / cpr, c = idx - row * cpr;
            int g = row >> 4, r = row & 15;
            uint4 v = *(const uint4*)(gWhh + (size_t)(g * HSZ + col + r) * HP + c * 8);
            *(uint4*)(sWhh + (size_t)row * LP + c * 8) = v;
        }
        if (isL0) {
            for (int idx = tid; idx < 48 * (IP / 8); idx += blockDim.x) {
                int row = idx >> 2, c = idx & 3;
                int g = row >> 4, r = row & 15;
                uint4 v = *(const uint4*)(P.Wih0 + (size_t)(g * HSZ + col + r) * IP + c * 8);
                *(uint4*)(sWih + (size_t)row * IP + c * 8) = v;
            }
        } else {
            for (int idx = tid; idx < 48 * cpr; idx += blockDim.x) {
                int row = idx / cpr, c = idx - row * cpr;
                int g = row >> 4, r = row & 15;
                uint4 v = *(const uint4*)(P.Wih1 + (size_t)(g * HSZ + col + r) * HP + c * 8);
                *(uint4*)(sWih + (size_t)row * LP + c * 8) = v;
            }
        }
    }
    __syncthreads();

    // hoisted biases (r/z fold both; n keeps b_hh inside the r* term)
    const float* b_ih = isL0 ? P.b_ih0 : P.b_ih1;
    const float* b_hh = isL0 ? P.b_hh0 : P.b_hh1;
    const float br   = b_ih[cn] + b_hh[cn];
    const float bz   = b_ih[HSZ + cn] + b_hh[HSZ + cn];
    const float bin_ = b_ih[2 * HSZ + cn];
    const float bhn  = b_hh[2 * HSZ + cn];

    // per-lane LDS B-tile bases (lane lr = tile column)
    const __bf16* hR = sWhh + (size_t)(0 * 16 + lr) * LP + half * 16;
    const __bf16* hZ = sWhh + (size_t)(1 * 16 + lr) * LP + half * 16;
    const __bf16* hN = sWhh + (size_t)(2 * 16 + lr) * LP + half * 16;
    const __bf16* iR = sWih + (size_t)(0 * 16 + lr) * ihPitch + half * 16;
    const __bf16* iZ = sWih + (size_t)(1 * 16 + lr) * ihPitch + half * 16;
    const __bf16* iN = sWih + (size_t)(2 * 16 + lr) * ihPitch + half * 16;

    unsigned barTarget = 0;
    for (int k = 0; k <= TSZ; ++k) {
        const bool active = isL0 ? (k < TSZ) : (k >= 1);   // block-uniform
        if (active) {
            const int s    = isL0 ? k : (k - 1);
            const int pin  = s & 1;
            const int pout = (s + 1) & 1;

            const __bf16* Ain;   int Kin;
            const __bf16* hprevb; const float* hprevf;
            __bf16* houtb;       float* houtf;
            if (isL0) {
                Ain = P.xb + (size_t)s * BSZ * IP;  Kin = IP;
                hprevb = P.h0b[pin];  hprevf = P.h0f[pin];
                houtb  = P.h0b[pout]; houtf  = P.h0f[pout];
            } else {
                Ain = P.h0b[pout];                  Kin = HP;   // out0[s], written at iter s
                hprevb = P.h1b[pin];  hprevf = P.h1f[pin];
                houtb  = P.h1b[pout]; houtf  = P.h1f[pout];
            }

            v8f accR = {}, accZ = {}, accGi = {}, accGh = {};
            // gi = A @ Wih^T   (r,z accumulate shared; n kept separate)
            gemm3(Ain + (size_t)(m0 + lr) * Kin + half * 16, iR, iZ, iN, Kin, accR, accZ, accGi);
            // gh = h_prev @ Whh^T
            gemm3(hprevb + (size_t)(m0 + lr) * HP + half * 16, hR, hZ, hN, HP, accR, accZ, accGh);

#pragma unroll
            for (int i = 0; i < 8; ++i) {
                const int   m    = m0 + half * 8 + i;   // C/D layout: lanes16-31 -> M=8+i
                const float r    = fast_sigmoid(accR[i] + br);
                const float z    = fast_sigmoid(accZ[i] + bz);
                const float n    = fast_tanh(accGi[i] + bin_ + r * (accGh[i] + bhn));
                const float hold = hprevf[(size_t)m * HSZ + cn];
                const float hnew = (1.0f - z) * n + z * hold;
                houtf[(size_t)m * HSZ + cn] = hnew;
                houtb[(size_t)m * HP  + cn] = (__bf16)hnew;
            }
        }

        // ---- device-wide epoch barrier ----
        barTarget += NBLK;
        __builtin_amdgcn_fence(__ATOMIC_RELEASE, "agent");   // my stores -> device scope
        __syncthreads();
        if (tid == 0) {
            __hip_atomic_fetch_add(P.bar, 1u, __ATOMIC_RELAXED, __HIP_MEMORY_SCOPE_AGENT);
            while (__hip_atomic_load(P.bar, __ATOMIC_RELAXED, __HIP_MEMORY_SCOPE_AGENT) < barTarget) {
                __builtin_amdgcn_s_sleep(1);
            }
        }
        __syncthreads();
        __builtin_amdgcn_fence(__ATOMIC_ACQUIRE, "agent");   // invalidate stale cached h
    }
}

// fp32 -> bf16 with K-padding (pad region zeroed)
__global__ void convert_pad_kernel(const float* __restrict__ src, __bf16* __restrict__ dst,
                                   int rows, int ksrc, int kdst)
{
    int idx = blockIdx.x * blockDim.x + threadIdx.x;
    if (idx >= rows * kdst) return;
    int r = idx / kdst;
    int k = idx - r * kdst;
    float v = (k < ksrc) ? src[(size_t)r * ksrc + k] : 0.0f;
    dst[idx] = (__bf16)v;
}

// x [B][T][I] fp32 -> xb [T][B][IP] bf16 (pad zeroed)
__global__ void convert_x_kernel(const float* __restrict__ x, __bf16* __restrict__ xb)
{
    int idx = blockIdx.x * blockDim.x + threadIdx.x;
    if (idx >= TSZ * BSZ * IP) return;
    int k  = idx & (IP - 1);
    int rb = idx >> 5;
    int b  = rb & (BSZ - 1);
    int t  = rb >> 6;
    float v = (k < ISZ) ? x[((size_t)b * TSZ + t) * ISZ + k] : 0.0f;
    xb[idx] = (__bf16)v;
}

__global__ void zero_kernel(uint32_t* __restrict__ p, int nwords)
{
    int i = blockIdx.x * blockDim.x + threadIdx.x;
    if (i < nwords) p[i] = 0u;
}

__global__ void copy_out_kernel(const float* __restrict__ h0, const float* __restrict__ h1,
                                float* __restrict__ out)
{
    int i = blockIdx.x * blockDim.x + threadIdx.x;
    if (i < BSZ * HSZ) {
        out[i]             = h0[i];
        out[BSZ * HSZ + i] = h1[i];
    }
}

extern "C" void kernel_launch(void* const* d_in, const int* in_sizes, int n_in,
                              void* d_out, int out_size, void* d_ws, size_t ws_size,
                              hipStream_t stream)
{
    (void)in_sizes; (void)n_in; (void)out_size; (void)ws_size;

    const float* x     = (const float*)d_in[0];
    const float* W_ih0 = (const float*)d_in[1];
    const float* W_hh0 = (const float*)d_in[2];
    const float* b_ih0 = (const float*)d_in[3];
    const float* b_hh0 = (const float*)d_in[4];
    const float* W_ih1 = (const float*)d_in[5];
    const float* W_hh1 = (const float*)d_in[6];
    const float* b_ih1 = (const float*)d_in[7];
    const float* b_hh1 = (const float*)d_in[8];

    char* ws = (char*)d_ws;
    size_t off = 0;
    auto carve = [&](size_t bytes) -> char* {
        char* p = ws + off;
        off += (bytes + 255) & ~(size_t)255;
        return p;
    };

    __bf16* Wih0b = (__bf16*)carve((size_t)G3 * IP * 2);
    __bf16* Whh0b = (__bf16*)carve((size_t)G3 * HP * 2);
    __bf16* Wih1b = (__bf16*)carve((size_t)G3 * HP * 2);
    __bf16* Whh1b = (__bf16*)carve((size_t)G3 * HP * 2);
    __bf16* xb    = (__bf16*)carve((size_t)TSZ * BSZ * IP * 2);

    char* hbase = ws + off;   // hidden buffers + barrier counter: one zero pass
    GruParams P;
    P.h0b[0] = (__bf16*)carve((size_t)BSZ * HP * 2);
    P.h0b[1] = (__bf16*)carve((size_t)BSZ * HP * 2);
    P.h1b[0] = (__bf16*)carve((size_t)BSZ * HP * 2);
    P.h1b[1] = (__bf16*)carve((size_t)BSZ * HP * 2);
    P.h0f[0] = (float*)carve((size_t)BSZ * HSZ * 4);
    P.h0f[1] = (float*)carve((size_t)BSZ * HSZ * 4);
    P.h1f[0] = (float*)carve((size_t)BSZ * HSZ * 4);
    P.h1f[1] = (float*)carve((size_t)BSZ * HSZ * 4);
    P.bar    = (unsigned*)carve(256);
    size_t hwords = ((ws + off) - hbase) / 4;

    P.xb = xb;
    P.Wih0 = Wih0b; P.Whh0 = Whh0b; P.Wih1 = Wih1b; P.Whh1 = Whh1b;
    P.b_ih0 = b_ih0; P.b_hh0 = b_hh0; P.b_ih1 = b_ih1; P.b_hh1 = b_hh1;

    // prep: zero hidden state + barrier, convert/pad weights and x to bf16
    zero_kernel<<<((int)hwords + 255) / 256, 256, 0, stream>>>((uint32_t*)hbase, (int)hwords);
    {
        int n;
        n = G3 * IP;
        convert_pad_kernel<<<(n + 255) / 256, 256, 0, stream>>>(W_ih0, Wih0b, G3, ISZ, IP);
        n = G3 * HP;
        convert_pad_kernel<<<(n + 255) / 256, 256, 0, stream>>>(W_hh0, Whh0b, G3, HSZ, HP);
        convert_pad_kernel<<<(n + 255) / 256, 256, 0, stream>>>(W_ih1, Wih1b, G3, HSZ, HP);
        convert_pad_kernel<<<(n + 255) / 256, 256, 0, stream>>>(W_hh1, Whh1b, G3, HSZ, HP);
        n = TSZ * BSZ * IP;
        convert_x_kernel<<<(n + 255) / 256, 256, 0, stream>>>(x, xb);
    }

    // persistent pipelined recurrence: 150 blocks, LDS-resident weights
    const size_t smemBytes = (size_t)2 * 48 * LP * sizeof(__bf16);  // 235,008 B < 320KB/WGP
    gru_persistent<<<NBLK, 128, smemBytes, stream>>>(P);

    // h_n = [h_last0, h_last1] : [2, 64, 1200] fp32 (TSZ even -> parity 0)
    copy_out_kernel<<<(BSZ * HSZ + 255) / 256, 256, 0, stream>>>(
        P.h0f[0], P.h1f[0], (float*)d_out);
}
